// EncoderProcesserDecoder_23493471109969
// MI455X (gfx1250) — matrix operand order repaired and checked by reading, not compiled, not run on hardware
//
#include <hip/hip_runtime.h>

// ---------------------------------------------------------------------------
// MeshGraphNet encoder-processor-decoder for MI455X (gfx1250).
//
// Roofline: ~380 GFLOP of 384x128 / 128x128 GEMMs over 300k edge/node rows,
// ~260MB working set (mostly L2-resident on the 192MB L2). bf16 WMMA
// (v_wmma_f32_16x16x32_bf16) makes compute time << gather/scatter time, so
// the kernels are organized around data movement: fused gather -> MLP -> LN
// -> atomic scatter-add -> residual, one 16x128 output tile per wave32.
// ---------------------------------------------------------------------------

typedef __bf16 bf16;
typedef __attribute__((ext_vector_type(16))) __bf16 bf16x16;
typedef __attribute__((ext_vector_type(8)))  __bf16 bf16x8;
typedef __attribute__((ext_vector_type(8)))  float  f32x8;

#define HID 128
#define WAVES 8          // waves per 256-thread block, each wave = 16 rows

struct MlpP {
  const unsigned short* w1f;  // W1 repacked into WMMA B-fragment layout (bf16)
  const unsigned short* w2f;  // W2 repacked
  const float* b1;
  const float* b2;
  const float* g;             // ln gamma
  const float* bt;            // ln beta
};

// ---- WMMA helpers ---------------------------------------------------------

__device__ __forceinline__ f32x8 wmma_bf16(bf16x16 a, bf16x16 b, f32x8 c) {
  // D = A(16x32) x B(32x16) + C(16x16 f32)
  return __builtin_amdgcn_wmma_f32_16x16x32_bf16(false, a, false, b, (short)0, c,
                                                 false, false);
}

// A fragment (16x32 bf16). ISA layout: lane holds row (lane&15);
// lanes 0-15 carry K {0..7,16..23}, lanes 16-31 carry K {8..15,24..31}.
// off = ktile_base + (lane>>4)*8 ; rp = row base pointer (f32, 16B aligned).
__device__ __forceinline__ bf16x16 afrag_global(const float* rp, int off) {
  float4 a0 = *(const float4*)(rp + off);
  float4 a1 = *(const float4*)(rp + off + 4);
  float4 c0 = *(const float4*)(rp + off + 16);
  float4 c1 = *(const float4*)(rp + off + 20);
  bf16x16 r;
  r[0]=(bf16)a0.x;  r[1]=(bf16)a0.y;  r[2]=(bf16)a0.z;  r[3]=(bf16)a0.w;
  r[4]=(bf16)a1.x;  r[5]=(bf16)a1.y;  r[6]=(bf16)a1.z;  r[7]=(bf16)a1.w;
  r[8]=(bf16)c0.x;  r[9]=(bf16)c0.y;  r[10]=(bf16)c0.z; r[11]=(bf16)c0.w;
  r[12]=(bf16)c1.x; r[13]=(bf16)c1.y; r[14]=(bf16)c1.z; r[15]=(bf16)c1.w;
  return r;
}

// Same A-fragment gather, but from a bf16 LDS row (16B-aligned chunks).
__device__ __forceinline__ bf16x16 afrag_lds(const bf16* rp, int off) {
  bf16x8 c0 = *(const bf16x8*)(rp + off);
  bf16x8 c1 = *(const bf16x8*)(rp + off + 16);
  bf16x16 r;
#pragma unroll
  for (int i = 0; i < 8; ++i) { r[i] = c0[i]; r[8 + i] = c1[i]; }
  return r;
}

// ---- weight repack: W[K][N] f32 -> per-fragment bf16 B layout -------------
// Fragment f = kt*ntiles + nt; lane stores 16 halfs contiguously so the GEMM
// loads each B fragment as two coalesced b128 loads.
__global__ void prep_wfrag(const float* __restrict__ W, unsigned short* __restrict__ out,
                           int K, int N, int ktiles, int ntiles) {
  int t = blockIdx.x * blockDim.x + threadIdx.x;
  int lane = t & 31;
  int f = t >> 5;
  if (f >= ktiles * ntiles) return;
  int kt = f / ntiles, nt = f % ntiles;
  int col = nt * 16 + (lane & 15);
  int kbase = kt * 32 + (lane >> 4) * 8;
  unsigned short* o = out + ((size_t)f * 32 + lane) * 16;
#pragma unroll
  for (int j = 0; j < 16; ++j) {
    int k = kbase + (j < 8 ? j : j + 8);
    float v = (k < K && col < N) ? W[(size_t)k * N + col] : 0.f;
    bf16 b = (bf16)v;
    o[j] = __builtin_bit_cast(unsigned short, b);
  }
}

__global__ void zero_kernel(float4* p, size_t n) {
  size_t i = (size_t)blockIdx.x * blockDim.x + threadIdx.x;
  size_t stride = (size_t)gridDim.x * blockDim.x;
  float4 z; z.x = z.y = z.z = z.w = 0.f;
  for (; i < n; i += stride) p[i] = z;
}

// ---- processor block MLP: y = LN(relu(concat(s0[i0],s1[i1],s2[i2])@W1+b1)@W2+b2)
//      resid += y ; optional agg[aggIdx[row]] += y (atomic f32) ---------------
__global__ __launch_bounds__(256) void block_mlp_kernel(
    MlpP p,
    const float* s0, const int* i0,
    const float* s1, const int* i1,
    const float* s2, const int* i2,
    float* resid,
    float* agg, const int* aggIdx,
    int ntiles) {
  __shared__ bf16 hbuf[WAVES][16 * 136];   // per-wave transpose staging
  const int lane = threadIdx.x & 31;
  const int wid = threadIdx.x >> 5;
  const int tile = blockIdx.x * WAVES + wid;
  if (tile >= ntiles) return;              // whole-wave exit keeps EXEC uniform
  const int row0 = tile * 16;
  const int h = lane >> 4;
  const int cl = lane & 15;

  const int ra = row0 + cl;                // this lane's A row
  const float* rp0 = s0 + (size_t)(i0 ? i0[ra] : ra) * HID;
  const float* rp1 = s1 + (size_t)(i1 ? i1[ra] : ra) * HID;
  const float* rp2 = s2 + (size_t)(i2 ? i2[ra] : ra) * HID;

  const f32x8 fz = {0.f, 0.f, 0.f, 0.f, 0.f, 0.f, 0.f, 0.f};
  f32x8 acc[8];
#pragma unroll
  for (int nt = 0; nt < 8; ++nt) acc[nt] = fz;

  // stage 1: [16x384] @ [384x128]
  const bf16x16* w1 = (const bf16x16*)p.w1f;
#pragma unroll
  for (int kt = 0; kt < 12; ++kt) {
    const float* rp = (kt < 4) ? rp0 : ((kt < 8) ? rp1 : rp2);
    bf16x16 a = afrag_global(rp, (kt & 3) * 32 + h * 8);
    const bf16x16* bp = w1 + (size_t)kt * 256 + lane;
#pragma unroll
    for (int nt = 0; nt < 8; ++nt)
      acc[nt] = wmma_bf16(a, bp[nt * 32], acc[nt]);
  }

  // bias + relu, transpose C-layout -> row-major bf16 in LDS
  bf16* my = hbuf[wid];
#pragma unroll
  for (int nt = 0; nt < 8; ++nt) {
    float bb = p.b1[nt * 16 + cl];
#pragma unroll
    for (int v = 0; v < 8; ++v) {
      float y = acc[nt][v] + bb;
      my[(h * 8 + v) * 136 + nt * 16 + cl] = (bf16)(y > 0.f ? y : 0.f);
    }
  }
  __builtin_amdgcn_wave_barrier();

  // stage 2: [16x128] @ [128x128]
  f32x8 acc2[8];
#pragma unroll
  for (int nt = 0; nt < 8; ++nt) acc2[nt] = fz;
  const bf16x16* w2 = (const bf16x16*)p.w2f;
  const bf16* arow = my + cl * 136;
#pragma unroll
  for (int kt = 0; kt < 4; ++kt) {
    bf16x16 a = afrag_lds(arow, kt * 32 + h * 8);
    const bf16x16* bp = w2 + (size_t)kt * 256 + lane;
#pragma unroll
    for (int nt = 0; nt < 8; ++nt)
      acc2[nt] = wmma_bf16(a, bp[nt * 32], acc2[nt]);
  }
#pragma unroll
  for (int nt = 0; nt < 8; ++nt) {
    float bb = p.b2[nt * 16 + cl];
#pragma unroll
    for (int v = 0; v < 8; ++v) acc2[nt][v] += bb;
  }

  // LayerNorm over 128 features: per-row sums across 8 nt then 16-lane halves
  float mean[8], rstd[8];
#pragma unroll
  for (int v = 0; v < 8; ++v) {
    float s = 0.f, q = 0.f;
#pragma unroll
    for (int nt = 0; nt < 8; ++nt) { float t = acc2[nt][v]; s += t; q += t * t; }
#pragma unroll
    for (int m = 1; m < 16; m <<= 1) {
      s += __shfl_xor(s, m, 32);
      q += __shfl_xor(q, m, 32);
    }
    float mu = s * (1.f / 128.f);
    mean[v] = mu;
    rstd[v] = rsqrtf(q * (1.f / 128.f) - mu * mu + 1e-5f);
  }

  int gidx[8];
#pragma unroll
  for (int v = 0; v < 8; ++v)
    gidx[v] = aggIdx ? aggIdx[row0 + h * 8 + v] : 0;

#pragma unroll
  for (int nt = 0; nt < 8; ++nt) {
    int col = nt * 16 + cl;
    float gg = p.g[col], bb = p.bt[col];
#pragma unroll
    for (int v = 0; v < 8; ++v) {
      int grow = row0 + h * 8 + v;
      float y = (acc2[nt][v] - mean[v]) * rstd[v] * gg + bb;
      if (agg) unsafeAtomicAdd(&agg[(size_t)gidx[v] * HID + col], y);  // segment_sum
      float* rr = resid + (size_t)grow * HID + col;                    // residual
      *rr = *rr + y;
    }
  }
}

// ---- encoder MLP: din (<=32, zero padded) -> 128 -> 128 -> LN, plain store --
__global__ __launch_bounds__(256) void encoder_mlp_kernel(
    MlpP p, const float* feat, int din, float* out, int ntiles) {
  __shared__ bf16 hbuf[WAVES][16 * 136];
  const int lane = threadIdx.x & 31;
  const int wid = threadIdx.x >> 5;
  const int tile = blockIdx.x * WAVES + wid;
  if (tile >= ntiles) return;
  const int row0 = tile * 16;
  const int h = lane >> 4;
  const int cl = lane & 15;
  const int ra = row0 + cl;

  const float* rp = feat + (size_t)ra * din;
  bf16x16 a;
#pragma unroll
  for (int j = 0; j < 8; ++j) {
    int k0 = h * 8 + j;
    int k1 = 16 + h * 8 + j;
    a[j]     = (bf16)(k0 < din ? rp[k0] : 0.f);
    a[8 + j] = (bf16)(k1 < din ? rp[k1] : 0.f);
  }

  const f32x8 fz = {0.f, 0.f, 0.f, 0.f, 0.f, 0.f, 0.f, 0.f};
  f32x8 acc[8];
#pragma unroll
  for (int nt = 0; nt < 8; ++nt) acc[nt] = fz;
  const bf16x16* w1 = (const bf16x16*)p.w1f;
  const bf16x16* bp1 = w1 + lane;
#pragma unroll
  for (int nt = 0; nt < 8; ++nt)
    acc[nt] = wmma_bf16(a, bp1[nt * 32], acc[nt]);

  bf16* my = hbuf[wid];
#pragma unroll
  for (int nt = 0; nt < 8; ++nt) {
    float bb = p.b1[nt * 16 + cl];
#pragma unroll
    for (int v = 0; v < 8; ++v) {
      float y = acc[nt][v] + bb;
      my[(h * 8 + v) * 136 + nt * 16 + cl] = (bf16)(y > 0.f ? y : 0.f);
    }
  }
  __builtin_amdgcn_wave_barrier();

  f32x8 acc2[8];
#pragma unroll
  for (int nt = 0; nt < 8; ++nt) acc2[nt] = fz;
  const bf16x16* w2 = (const bf16x16*)p.w2f;
  const bf16* arow = my + cl * 136;
#pragma unroll
  for (int kt = 0; kt < 4; ++kt) {
    bf16x16 a2 = afrag_lds(arow, kt * 32 + h * 8);
    const bf16x16* bp = w2 + (size_t)kt * 256 + lane;
#pragma unroll
    for (int nt = 0; nt < 8; ++nt)
      acc2[nt] = wmma_bf16(a2, bp[nt * 32], acc2[nt]);
  }
#pragma unroll
  for (int nt = 0; nt < 8; ++nt) {
    float bb = p.b2[nt * 16 + cl];
#pragma unroll
    for (int v = 0; v < 8; ++v) acc2[nt][v] += bb;
  }

  float mean[8], rstd[8];
#pragma unroll
  for (int v = 0; v < 8; ++v) {
    float s = 0.f, q = 0.f;
#pragma unroll
    for (int nt = 0; nt < 8; ++nt) { float t = acc2[nt][v]; s += t; q += t * t; }
#pragma unroll
    for (int m = 1; m < 16; m <<= 1) {
      s += __shfl_xor(s, m, 32);
      q += __shfl_xor(q, m, 32);
    }
    float mu = s * (1.f / 128.f);
    mean[v] = mu;
    rstd[v] = rsqrtf(q * (1.f / 128.f) - mu * mu + 1e-5f);
  }

#pragma unroll
  for (int nt = 0; nt < 8; ++nt) {
    int col = nt * 16 + cl;
    float gg = p.g[col], bb = p.bt[col];
#pragma unroll
    for (int v = 0; v < 8; ++v) {
      int grow = row0 + h * 8 + v;
      out[(size_t)grow * HID + col] = (acc2[nt][v] - mean[v]) * rstd[v] * gg + bb;
    }
  }
}

// ---- decoder: out[40000x4] = relu(x[mask] @ W1 + b1) @ W2 + b2 (no LN) ------
__global__ __launch_bounds__(256) void decoder_kernel(
    const unsigned short* w1f, const unsigned short* w2f,
    const float* b1, const float* b2,
    const float* x, const int* mask, float* out, int ntiles) {
  __shared__ bf16 hbuf[WAVES][16 * 136];
  const int lane = threadIdx.x & 31;
  const int wid = threadIdx.x >> 5;
  const int tile = blockIdx.x * WAVES + wid;
  if (tile >= ntiles) return;
  const int row0 = tile * 16;
  const int h = lane >> 4;
  const int cl = lane & 15;
  const int ra = row0 + cl;

  const float* rp = x + (size_t)mask[ra] * HID;
  const f32x8 fz = {0.f, 0.f, 0.f, 0.f, 0.f, 0.f, 0.f, 0.f};
  f32x8 acc[8];
#pragma unroll
  for (int nt = 0; nt < 8; ++nt) acc[nt] = fz;
  const bf16x16* w1 = (const bf16x16*)w1f;
#pragma unroll
  for (int kt = 0; kt < 4; ++kt) {
    bf16x16 a = afrag_global(rp, kt * 32 + h * 8);
    const bf16x16* bp = w1 + (size_t)kt * 256 + lane;
#pragma unroll
    for (int nt = 0; nt < 8; ++nt)
      acc[nt] = wmma_bf16(a, bp[nt * 32], acc[nt]);
  }

  bf16* my = hbuf[wid];
#pragma unroll
  for (int nt = 0; nt < 8; ++nt) {
    float bb = b1[nt * 16 + cl];
#pragma unroll
    for (int v = 0; v < 8; ++v) {
      float y = acc[nt][v] + bb;
      my[(h * 8 + v) * 136 + nt * 16 + cl] = (bf16)(y > 0.f ? y : 0.f);
    }
  }
  __builtin_amdgcn_wave_barrier();

  f32x8 acc2 = fz;                         // single N tile (OUT=4 padded to 16)
  const bf16x16* w2 = (const bf16x16*)w2f;
  const bf16* arow = my + cl * 136;
#pragma unroll
  for (int kt = 0; kt < 4; ++kt) {
    bf16x16 a2 = afrag_lds(arow, kt * 32 + h * 8);
    acc2 = wmma_bf16(a2, w2[(size_t)kt * 32 + lane], acc2);
  }
  if (cl < 4) {
    float bb = b2[cl];
#pragma unroll
    for (int v = 0; v < 8; ++v)
      out[(size_t)(row0 + h * 8 + v) * 4 + cl] = acc2[v] + bb;
  }
}

// ---------------------------------------------------------------------------
// Host side
// ---------------------------------------------------------------------------
extern "C" void kernel_launch(void* const* d_in, const int* in_sizes, int n_in,
                              void* d_out, int out_size, void* d_ws, size_t ws_size,
                              hipStream_t stream) {
  (void)in_sizes; (void)n_in; (void)out_size; (void)ws_size;
  const int NN = 50000, EM = 200000, EW = 100000, NMASK = 40000;
  const int NT_N = NN / 16, NT_EM = EM / 16, NT_EW = EW / 16, NT_MASK = NMASK / 16;

  const float* node_feat = (const float*)d_in[0];
  const float* em_attr   = (const float*)d_in[1];
  const float* ew_attr   = (const float*)d_in[2];
  const int*   eidx      = (const int*)d_in[3];
  const int*   ewidx     = (const int*)d_in[4];
  const int*   mask      = (const int*)d_in[5];
  const int *sm = eidx, *rm = eidx + EM;
  const int *sw = ewidx, *rw = ewidx + EW;

  // --- workspace carve (256B aligned) ---
  char* ws = (char*)d_ws;
  size_t off = 0;
  auto carve = [&](size_t bytes) -> void* {
    void* p = ws + off;
    off += (bytes + 255) & ~(size_t)255;
    return p;
  };
  float* x    = (float*)carve((size_t)NN * HID * 4);
  float* em   = (float*)carve((size_t)EM * HID * 4);
  float* ew   = (float*)carve((size_t)EW * HID * 4);
  float* aggm = (float*)carve((size_t)NN * HID * 4);
  float* aggw = (float*)carve((size_t)NN * HID * 4);

  // params flattened jax-pytree style: top-level insertion order, nested dicts
  // by sorted key. MLP w/ LN: [b1,b2,ln_b,ln_g,w1,w2]; dec: [b1,b2,w1,w2].
  auto P = [&](int i) -> const float* { return (const float*)d_in[i]; };

  auto prep = [&](const float* W, int K, int Ncols, int kt, int nt) -> unsigned short* {
    unsigned short* f = (unsigned short*)carve((size_t)kt * nt * 512 * 2);
    int threads = kt * nt * 32;
    prep_wfrag<<<(threads + 255) / 256, 256, 0, stream>>>(W, f, K, Ncols, kt, nt);
    return f;
  };

  MlpP blk[8][3];
  for (int l = 0; l < 8; ++l) {
    for (int m = 0; m < 3; ++m) {          // sorted: emb, ewb, nb
      int b = 6 + l * 18 + m * 6;
      MlpP p;
      p.b1 = P(b + 0); p.b2 = P(b + 1); p.bt = P(b + 2); p.g = P(b + 3);
      p.w1f = prep(P(b + 4), 384, HID, 12, 8);
      p.w2f = prep(P(b + 5), HID, HID, 4, 8);
      blk[l][m] = p;
    }
  }
  const float* dec_b1 = P(150);
  const float* dec_b2 = P(151);
  unsigned short* dec_w1f = prep(P(152), HID, HID, 4, 8);
  unsigned short* dec_w2f = prep(P(153), HID, 4, 4, 1);

  auto encP = [&](int b, int din) -> MlpP {
    MlpP p;
    p.b1 = P(b + 0); p.b2 = P(b + 1); p.bt = P(b + 2); p.g = P(b + 3);
    p.w1f = prep(P(b + 4), din, HID, 1, 8);
    p.w2f = prep(P(b + 5), HID, HID, 4, 8);
    return p;
  };
  MlpP enc_emb = encP(154, 8);
  MlpP enc_ewb = encP(160, 4);
  MlpP enc_nb  = encP(166, 12);

  // --- encoder ---
  encoder_mlp_kernel<<<(NT_N  + WAVES - 1) / WAVES, 256, 0, stream>>>(enc_nb,  node_feat, 12, x,  NT_N);
  encoder_mlp_kernel<<<(NT_EM + WAVES - 1) / WAVES, 256, 0, stream>>>(enc_emb, em_attr,    8, em, NT_EM);
  encoder_mlp_kernel<<<(NT_EW + WAVES - 1) / WAVES, 256, 0, stream>>>(enc_ewb, ew_attr,    4, ew, NT_EW);

  // --- processor: 8 GN blocks ---
  size_t aggn4 = (size_t)NN * HID / 4;
  for (int l = 0; l < 8; ++l) {
    zero_kernel<<<2048, 256, 0, stream>>>((float4*)aggm, aggn4);
    zero_kernel<<<2048, 256, 0, stream>>>((float4*)aggw, aggn4);
    block_mlp_kernel<<<(NT_EM + WAVES - 1) / WAVES, 256, 0, stream>>>(
        blk[l][0], x, sm, x, rm, em, (const int*)nullptr, em, aggm, rm, NT_EM);
    block_mlp_kernel<<<(NT_EW + WAVES - 1) / WAVES, 256, 0, stream>>>(
        blk[l][1], x, sw, x, rw, ew, (const int*)nullptr, ew, aggw, rw, NT_EW);
    block_mlp_kernel<<<(NT_N + WAVES - 1) / WAVES, 256, 0, stream>>>(
        blk[l][2], x, (const int*)nullptr, aggm, (const int*)nullptr,
        aggw, (const int*)nullptr, x, (float*)nullptr, (const int*)nullptr, NT_N);
  }

  // --- decoder ---
  decoder_kernel<<<(NT_MASK + WAVES - 1) / WAVES, 256, 0, stream>>>(
      dec_w1f, dec_w2f, dec_b1, dec_b2, x, mask, (float*)d_out, NT_MASK);
}
